// GraphnetDynedge_10479720202281
// MI455X (gfx1250) — matrix operand order repaired and checked by reading, not compiled, not run on hardware
//
#include <hip/hip_runtime.h>

// ---------------------------------------------------------------- types
typedef __bf16 bf16_t;
typedef bf16_t v16bf __attribute__((ext_vector_type(16)));
typedef bf16_t v8bf  __attribute__((ext_vector_type(8)));
typedef float  v8f   __attribute__((ext_vector_type(8)));

union Frag { v16bf v; v8bf h[2]; };

__device__ static inline v8f wmma_bf16(v16bf a, v16bf b, v8f c) {
  return __builtin_amdgcn_wmma_f32_16x16x32_bf16(false, a, false, b, (short)0, c,
                                                 false, false);
}

// A-matrix 16x32 bf16 fragment from row-major tile (ld in elements).
// lane<16: M=lane, K = {0..7, 16..23}; lane>=16: M=lane-16, K = {8..15, 24..31}
__device__ static inline v16bf load_a_lds(const bf16_t* base, int ld) {
  int lane = threadIdx.x & 31;
  const bf16_t* p = base + (lane & 15) * ld + ((lane >> 4) << 3);
  Frag f;
  f.h[0] = *(const v8bf*)(p);
  f.h[1] = *(const v8bf*)(p + 16);
  return f.v;
}

// B-matrix 32x16 bf16 fragment from column-major weights Wt[n][k] (ld = K-pad).
// lane<16: N=lane, K = {0..7, 16..23}; lane>=16: N=lane-16, K = {8..15, 24..31}
__device__ static inline v16bf load_b_glob(const bf16_t* base, int ld) {
  int lane = threadIdx.x & 31;
  const bf16_t* p = base + (lane & 15) * ld + ((lane >> 4) << 3);
  Frag f;
  f.h[0] = *(const v8bf*)(p);
  f.h[1] = *(const v8bf*)(p + 16);
  return f.v;
}

__device__ static inline v8f vzero8() {
  v8f z = {0.f, 0.f, 0.f, 0.f, 0.f, 0.f, 0.f, 0.f};
  return z;
}

#define N_NODES 8192
#define FDIM 256

// ---------------------------------------------------------------- weight pack
// W: fp32 [K][N] row-major -> Wt: bf16 [N][Kpad] column-major, zero padded K
__global__ __launch_bounds__(256) void pack_wt_kernel(
    const float* __restrict__ w, bf16_t* __restrict__ wt, int K, int N, int Kpad) {
  int idx = blockIdx.x * 256 + threadIdx.x;
  if (idx >= N * Kpad) return;
  int n = idx / Kpad;
  int k = idx - n * Kpad;
  wt[idx] = (k < K) ? (bf16_t)w[k * N + n] : (bf16_t)0.0f;
}

// ---------------------------------------------------------------- conv1
// edges from edge_index (src list, dst = e/4). e = [xi(5), xj-xi(5)] pad->32
// MLP: 32->128 relu -> 128->256 relu ; sum groups of 4 edges -> x1
__global__ __launch_bounds__(256) void conv1_kernel(
    const float* __restrict__ x0, const int* __restrict__ src,
    const bf16_t* __restrict__ wt1, const float* __restrict__ b1,
    const bf16_t* __restrict__ wt2, const float* __restrict__ b2,
    float* __restrict__ xout) {
  __shared__ bf16_t A1[64][32];
  __shared__ bf16_t A2[64][128];
  const int t = threadIdx.x;
  const int nb = blockIdx.x * 16;   // first node of block
  const int lane = t & 31, nl = lane & 15, hl = lane >> 4;
  const int wave = t >> 5;

  // phase 0: build edge features
  {
    int e = t >> 2;            // 0..63
    int p = t & 3;             // col quarter (8 cols each)
    int ge = blockIdx.x * 64 + e;
    int i = ge >> 2;
    int j = src[ge];
    const float* xi = x0 + i * 5;
    const float* xj = x0 + j * 5;
#pragma unroll
    for (int c = 0; c < 8; ++c) {
      int col = p * 8 + c;
      float v;
      if (col < 5)        v = xi[col];
      else if (col < 10)  v = xj[col - 5] - xi[col - 5];
      else                v = 0.0f;
      A1[e][col] = (bf16_t)v;
    }
  }
  __syncthreads();

  // phase 1: 64x32 @ 32x128  (8 n-tiles, one per wave; single k-step)
  {
    int nt = wave;
    v8f acc[4];
#pragma unroll
    for (int m = 0; m < 4; ++m) acc[m] = vzero8();
    v16bf bf = load_b_glob(wt1 + (size_t)(nt * 16) * 32, 32);
#pragma unroll
    for (int m = 0; m < 4; ++m) {
      v16bf af = load_a_lds(&A1[m * 16][0], 32);
      acc[m] = wmma_bf16(af, bf, acc[m]);
    }
    int col = nt * 16 + nl;
    float bias = b1[col];
#pragma unroll
    for (int m = 0; m < 4; ++m)
#pragma unroll
      for (int r = 0; r < 8; ++r) {
        float v = acc[m][r] + bias;
        if (v < 0.f) v = 0.f;
        A2[m * 16 + r + 8 * hl][col] = (bf16_t)v;
      }
  }
  __syncthreads();

  // phase 2: 64x128 @ 128x256 ; relu ; sum 4-edge groups -> 16 node rows
  for (int nt = wave; nt < 16; nt += 8) {
    v8f acc[4];
#pragma unroll
    for (int m = 0; m < 4; ++m) acc[m] = vzero8();
    for (int kt = 0; kt < 4; ++kt) {
      v16bf bf = load_b_glob(wt2 + (size_t)(nt * 16) * 128 + kt * 32, 128);
#pragma unroll
      for (int m = 0; m < 4; ++m) {
        v16bf af = load_a_lds(&A2[m * 16][kt * 32], 128);
        acc[m] = wmma_bf16(af, bf, acc[m]);
      }
    }
    int col = nt * 16 + nl;
    float bias = b2[col];
#pragma unroll
    for (int m = 0; m < 4; ++m) {
      float s[8];
#pragma unroll
      for (int r = 0; r < 8; ++r) {
        float v = acc[m][r] + bias;
        s[r] = v < 0.f ? 0.f : v;
      }
      float sa = s[0] + s[1] + s[2] + s[3];
      float sb = s[4] + s[5] + s[6] + s[7];
      int node = nb + m * 4 + hl * 2;   // rows 0-3 -> node, 4-7 -> node+1 per half
      xout[node * FDIM + col] = sa;
      xout[(node + 1) * FDIM + col] = sb;
    }
  }
}

// ---------------------------------------------------------------- kNN (k=4)
__global__ __launch_bounds__(128) void knn_kernel(const float* __restrict__ x,
                                                  int* __restrict__ nbr) {
  __shared__ float xg[128][260];
  const int g = blockIdx.x;
  const int t = threadIdx.x;
  const int base = g * 128;
  for (int idx = t; idx < 128 * 256; idx += 128) {
    int r = idx >> 8, c = idx & 255;
    xg[r][c] = x[(size_t)(base + r) * FDIM + c];
  }
  __syncthreads();

  const int i = t;
  float bd0 = 3.0e38f, bd1 = 3.0e38f, bd2 = 3.0e38f, bd3 = 3.0e38f;
  int bi0 = 0, bi1 = 0, bi2 = 0, bi3 = 0;
  for (int j = 0; j < 128; ++j) {
    if (j == i) continue;
    float d = 0.f;
#pragma unroll 4
    for (int c = 0; c < 256; c += 4) {
      float4 a = *(const float4*)&xg[i][c];
      float4 b = *(const float4*)&xg[j][c];
      float dx = a.x - b.x, dy = a.y - b.y, dz = a.z - b.z, dw = a.w - b.w;
      d += dx * dx + dy * dy + dz * dz + dw * dw;
    }
    if (d < bd3) {
      if (d < bd2) {
        bd3 = bd2; bi3 = bi2;
        if (d < bd1) {
          bd2 = bd1; bi2 = bi1;
          if (d < bd0) { bd1 = bd0; bi1 = bi0; bd0 = d; bi0 = j; }
          else         { bd1 = d;  bi1 = j; }
        } else         { bd2 = d;  bi2 = j; }
      } else           { bd3 = d;  bi3 = j; }
    }
  }
  int* o = nbr + (size_t)(base + i) * 4;
  o[0] = base + bi0; o[1] = base + bi1; o[2] = base + bi2; o[3] = base + bi3;
}

// ---------------------------------------------------------------- dyn conv
// e = [xi(256), xj-xi(256)] ; MLP: 512->336 relu -> 336->256 relu ; sum k=4
__global__ __launch_bounds__(256) void dyn_conv_kernel(
    const float* __restrict__ x, const int* __restrict__ nbr,
    const bf16_t* __restrict__ wt1, const float* __restrict__ b1,
    const bf16_t* __restrict__ wt2, const float* __restrict__ b2,
    float* __restrict__ xout) {
  __shared__ bf16_t A1[64][512];   // 64 edges x 512 in
  __shared__ bf16_t A2[64][352];   // 64 edges x 336 hidden (padded to 352)
  const int t = threadIdx.x;
  const int nb = blockIdx.x * 16;
  const int lane = t & 31, nl = lane & 15, hl = lane >> 4;
  const int wave = t >> 5;

  // phase 0: gather edge features (4 threads per edge row, 128 cols each)
  {
    int e = t >> 2;
    int p = t & 3;
    int i = nb + (e >> 2);
    int j = nbr[i * 4 + (e & 3)];
    const float* xi = x + (size_t)i * FDIM;
    const float* xj = x + (size_t)j * FDIM;
    int c0 = p * 128;
    if (p < 2) {
      for (int c = 0; c < 128; c += 4) {
        float4 v = *(const float4*)(xi + c0 + c);
        A1[e][c0 + c + 0] = (bf16_t)v.x;
        A1[e][c0 + c + 1] = (bf16_t)v.y;
        A1[e][c0 + c + 2] = (bf16_t)v.z;
        A1[e][c0 + c + 3] = (bf16_t)v.w;
      }
    } else {
      int d0 = c0 - 256;
      for (int c = 0; c < 128; c += 4) {
        float4 a = *(const float4*)(xi + d0 + c);
        float4 b = *(const float4*)(xj + d0 + c);
        A1[e][c0 + c + 0] = (bf16_t)(b.x - a.x);
        A1[e][c0 + c + 1] = (bf16_t)(b.y - a.y);
        A1[e][c0 + c + 2] = (bf16_t)(b.z - a.z);
        A1[e][c0 + c + 3] = (bf16_t)(b.w - a.w);
      }
    }
    // zero the K-pad columns of A2 (336..351)
    for (int q = t; q < 64 * 16; q += 256)
      A2[q >> 4][336 + (q & 15)] = (bf16_t)0.0f;
  }
  __syncthreads();

  // phase 1: 64x512 @ 512x336 (21 n-tiles over 8 waves, 16 k-steps)
  for (int nt = wave; nt < 21; nt += 8) {
    v8f acc[4];
#pragma unroll
    for (int m = 0; m < 4; ++m) acc[m] = vzero8();
    for (int kt = 0; kt < 16; ++kt) {
      v16bf bf = load_b_glob(wt1 + (size_t)(nt * 16) * 512 + kt * 32, 512);
#pragma unroll
      for (int m = 0; m < 4; ++m) {
        v16bf af = load_a_lds(&A1[m * 16][kt * 32], 512);
        acc[m] = wmma_bf16(af, bf, acc[m]);
      }
    }
    int col = nt * 16 + nl;
    float bias = b1[col];
#pragma unroll
    for (int m = 0; m < 4; ++m)
#pragma unroll
      for (int r = 0; r < 8; ++r) {
        float v = acc[m][r] + bias;
        if (v < 0.f) v = 0.f;
        A2[m * 16 + r + 8 * hl][col] = (bf16_t)v;
      }
  }
  __syncthreads();

  // phase 2: 64x352 @ 352x256 ; relu ; sum 4-edge groups -> node rows
  for (int nt = wave; nt < 16; nt += 8) {
    v8f acc[4];
#pragma unroll
    for (int m = 0; m < 4; ++m) acc[m] = vzero8();
    for (int kt = 0; kt < 11; ++kt) {
      v16bf bf = load_b_glob(wt2 + (size_t)(nt * 16) * 352 + kt * 32, 352);
#pragma unroll
      for (int m = 0; m < 4; ++m) {
        v16bf af = load_a_lds(&A2[m * 16][kt * 32], 352);
        acc[m] = wmma_bf16(af, bf, acc[m]);
      }
    }
    int col = nt * 16 + nl;
    float bias = b2[col];
#pragma unroll
    for (int m = 0; m < 4; ++m) {
      float s[8];
#pragma unroll
      for (int r = 0; r < 8; ++r) {
        float v = acc[m][r] + bias;
        s[r] = v < 0.f ? 0.f : v;
      }
      float sa = s[0] + s[1] + s[2] + s[3];
      float sb = s[4] + s[5] + s[6] + s[7];
      int node = nb + m * 4 + hl * 2;
      xout[(size_t)node * FDIM + col] = sa;
      xout[(size_t)(node + 1) * FDIM + col] = sb;
    }
  }
}

// ---------------------------------------------------------------- head
__global__ __launch_bounds__(256) void head_kernel(
    const float* __restrict__ x0, const float* __restrict__ x1,
    const float* __restrict__ x2, const float* __restrict__ x3,
    const float* __restrict__ x4, const float* __restrict__ hw,
    const float* __restrict__ hb, float* __restrict__ out) {
  int n = blockIdx.x * 256 + threadIdx.x;
  if (n >= N_NODES) return;
  float a0 = hb[0], a1 = hb[1];
#pragma unroll
  for (int c = 0; c < 5; ++c) {
    float f = x0[n * 5 + c];
    a0 += f * hw[c * 2 + 0];
    a1 += f * hw[c * 2 + 1];
  }
  const float* xs[4] = {x1, x2, x3, x4};
  int row = 5;
  for (int s = 0; s < 4; ++s) {
    const float* xp = xs[s] + (size_t)n * FDIM;
    for (int c = 0; c < FDIM; ++c) {
      float f = xp[c];
      a0 += f * hw[(row + c) * 2 + 0];
      a1 += f * hw[(row + c) * 2 + 1];
    }
    row += FDIM;
  }
  out[n * 2 + 0] = a0;
  out[n * 2 + 1] = a1;
}

// ---------------------------------------------------------------- launch
extern "C" void kernel_launch(void* const* d_in, const int* in_sizes, int n_in,
                              void* d_out, int out_size, void* d_ws, size_t ws_size,
                              hipStream_t stream) {
  (void)in_sizes; (void)n_in; (void)out_size; (void)ws_size;
  const float* x0  = (const float*)d_in[0];
  const int*   src = (const int*)d_in[1];           // edge_index row 0
  // d_in[2] = batch (graphs are contiguous blocks of 128 by construction)
  const float* W1[4]; const float* B1[4]; const float* W2[4]; const float* B2[4];
  for (int c = 0; c < 4; ++c) {
    W1[c] = (const float*)d_in[3 + c * 4 + 0];
    B1[c] = (const float*)d_in[3 + c * 4 + 1];
    W2[c] = (const float*)d_in[3 + c * 4 + 2];
    B2[c] = (const float*)d_in[3 + c * 4 + 3];
  }
  const float* hw = (const float*)d_in[19];
  const float* hb = (const float*)d_in[20];

  // workspace carve-out (256B aligned)
  char* ws = (char*)d_ws;
  size_t off = 0;
  auto carve = [&](size_t bytes) {
    off = (off + 255) & ~(size_t)255;
    char* p = ws + off;
    off += bytes;
    return p;
  };
  const size_t xbytes = (size_t)N_NODES * FDIM * sizeof(float);
  float* x1 = (float*)carve(xbytes);
  float* x2 = (float*)carve(xbytes);
  float* x3 = (float*)carve(xbytes);
  float* x4 = (float*)carve(xbytes);
  int* nbr  = (int*)carve((size_t)N_NODES * 4 * sizeof(int));
  bf16_t* wt1_0 = (bf16_t*)carve((size_t)128 * 32 * 2);
  bf16_t* wt2_0 = (bf16_t*)carve((size_t)256 * 128 * 2);
  bf16_t* wt1_c[3]; bf16_t* wt2_c[3];
  for (int c = 0; c < 3; ++c) {
    wt1_c[c] = (bf16_t*)carve((size_t)336 * 512 * 2);
    wt2_c[c] = (bf16_t*)carve((size_t)256 * 352 * 2);
  }

  // pack weights (fp32 row-major -> bf16 column-major, K padded)
  pack_wt_kernel<<<(128 * 32 + 255) / 256, 256, 0, stream>>>(W1[0], wt1_0, 10, 128, 32);
  pack_wt_kernel<<<(256 * 128 + 255) / 256, 256, 0, stream>>>(W2[0], wt2_0, 128, 256, 128);
  for (int c = 0; c < 3; ++c) {
    pack_wt_kernel<<<(336 * 512 + 255) / 256, 256, 0, stream>>>(W1[c + 1], wt1_c[c], 512, 336, 512);
    pack_wt_kernel<<<(256 * 352 + 255) / 256, 256, 0, stream>>>(W2[c + 1], wt2_c[c], 336, 256, 352);
  }

  // conv1 (precomputed edges)
  conv1_kernel<<<N_NODES / 16, 256, 0, stream>>>(x0, src, wt1_0, B1[0], wt2_0, B2[0], x1);

  // dynamic convs 2..4
  float* xin = x1;
  float* outs[3] = {x2, x3, x4};
  for (int c = 0; c < 3; ++c) {
    knn_kernel<<<64, 128, 0, stream>>>(xin, nbr);
    dyn_conv_kernel<<<N_NODES / 16, 256, 0, stream>>>(xin, nbr, wt1_c[c], B1[c + 1],
                                                      wt2_c[c], B2[c + 1], outs[c]);
    xin = outs[c];
  }

  // head
  head_kernel<<<N_NODES / 256, 256, 0, stream>>>(x0, x1, x2, x3, x4, hw, hb,
                                                 (float*)d_out);
}
